// HREncoder_23811298689130
// MI455X (gfx1250) — matrix operand order repaired and checked by reading, not compiled, not run on hardware
//
#include <hip/hip_runtime.h>
#include <hip/hip_bf16.h>

// ---------------------------------------------------------------------------
// HREncoder on MI455X (gfx1250): 2-layer bi-LSTM recurrence + VAE heads.
// All GEMMs run on v_wmma_f32_16x16x32_bf16 (f32 accum). Weights converted to
// bf16 once (resident in 192MB L2 across the 255 sequential steps); h-state
// kept bf16 (GEMM operand only), c-state and nonlinearities in f32.
// ---------------------------------------------------------------------------

typedef __attribute__((ext_vector_type(16))) __bf16 v16bf;
typedef __attribute__((ext_vector_type(8)))  __bf16 bf16x8;
typedef __attribute__((ext_vector_type(8)))  float  v8f;

#define B_  64
#define S_  256
#define E_  640
#define H_  512
#define NG  2048   // 4*H (gates)
#define ZD  4096   // L*2*2*H

__device__ __forceinline__ float sigmoidf_(float x) {
    return 1.0f / (1.0f + __expf(-x));
}

// ---- WMMA fragment loaders ------------------------------------------------
// A fragment (16x32, 16-bit): lane holds row m (lane%16). Element e holds
// K = kskew + e, kskew = kbase + (lane>=16 ? 16 : 0).
// => one contiguous 16-element run at [kskew, kskew+16).
__device__ __forceinline__ v16bf a_frag_f32(const float* row, int kskew) {
    const float4* p0 = reinterpret_cast<const float4*>(row + kskew);
    const float4* p1 = reinterpret_cast<const float4*>(row + kskew + 8);
    float4 x0 = p0[0], x1 = p0[1], x2 = p1[0], x3 = p1[1];
    v16bf r;
    r[0]=(__bf16)x0.x;  r[1]=(__bf16)x0.y;  r[2]=(__bf16)x0.z;  r[3]=(__bf16)x0.w;
    r[4]=(__bf16)x1.x;  r[5]=(__bf16)x1.y;  r[6]=(__bf16)x1.z;  r[7]=(__bf16)x1.w;
    r[8]=(__bf16)x2.x;  r[9]=(__bf16)x2.y;  r[10]=(__bf16)x2.z; r[11]=(__bf16)x2.w;
    r[12]=(__bf16)x3.x; r[13]=(__bf16)x3.y; r[14]=(__bf16)x3.z; r[15]=(__bf16)x3.w;
    return r;
}

__device__ __forceinline__ v16bf a_frag_bf16(const __bf16* row, int kskew) {
    bf16x8 lo = *reinterpret_cast<const bf16x8*>(row + kskew);
    bf16x8 hi = *reinterpret_cast<const bf16x8*>(row + kskew + 8);
    v16bf r;
#pragma unroll
    for (int e = 0; e < 8; ++e) { r[e] = lo[e]; r[8 + e] = hi[e]; }
    return r;
}

// B fragment (32x16, 16-bit): lane holds column n (lane%16). Element e holds
// K = kbase + (lane>=16 ? 16 : 0) + e  => one contiguous 16-element run.
// We feed it rows of W stored [N, K] (computing X @ W^T).
__device__ __forceinline__ v16bf b_frag_bf16(const __bf16* wrow, int kskew16) {
    bf16x8 lo = *reinterpret_cast<const bf16x8*>(wrow + kskew16);
    bf16x8 hi = *reinterpret_cast<const bf16x8*>(wrow + kskew16 + 8);
    v16bf r;
#pragma unroll
    for (int e = 0; e < 8; ++e) { r[e] = lo[e]; r[8 + e] = hi[e]; }
    return r;
}

__device__ __forceinline__ v16bf b_frag_f32(const float* wrow, int kskew16) {
    const float4* p = reinterpret_cast<const float4*>(wrow + kskew16);
    float4 x0 = p[0], x1 = p[1], x2 = p[2], x3 = p[3];
    v16bf r;
    r[0]=(__bf16)x0.x;  r[1]=(__bf16)x0.y;  r[2]=(__bf16)x0.z;  r[3]=(__bf16)x0.w;
    r[4]=(__bf16)x1.x;  r[5]=(__bf16)x1.y;  r[6]=(__bf16)x1.z;  r[7]=(__bf16)x1.w;
    r[8]=(__bf16)x2.x;  r[9]=(__bf16)x2.y;  r[10]=(__bf16)x2.z; r[11]=(__bf16)x2.w;
    r[12]=(__bf16)x3.x; r[13]=(__bf16)x3.y; r[14]=(__bf16)x3.z; r[15]=(__bf16)x3.w;
    return r;
}

__device__ __forceinline__ v8f wmma_bf16(v16bf a, v16bf b, v8f c) {
    return __builtin_amdgcn_wmma_f32_16x16x32_bf16(
        /*neg_a=*/false, a, /*neg_b=*/false, b,
        /*c_mod=*/(short)0, c, /*reuse_a=*/false, /*reuse_b=*/false);
}

// ---------------------------------------------------------------------------
// One bidirectional LSTM layer step (both directions). 256 waves:
//   wave -> (dir d, M-tile mt in 0..3, H-tile ht in 0..31)
// Each wave computes the four gate tiles (i,f,g,o) sharing one A fragment,
// then fuses the cell nonlinearity and the h/c state update.
// ---------------------------------------------------------------------------
template <bool X_IS_F32>
__global__ __launch_bounds__(256)
void lstm_step_kernel(const float*  __restrict__ Xf,
                      const __bf16* __restrict__ Xb,
                      int ldx, int Kx,
                      const __bf16* __restrict__ hprev, int hprev_dir_off, int hprev_ld,
                      const __bf16* __restrict__ wih,   // [2][NG][Kx]  bf16
                      const __bf16* __restrict__ whh,   // [2][NG][H]   bf16
                      const float*  __restrict__ bias,  // [2][NG]      f32 (b_ih+b_hh)
                      float*        __restrict__ cstate,// [2][B][H]    f32, in-place
                      __bf16*       __restrict__ hnew, int hnew_dir_off, int hnew_ld)
{
    const int w    = (blockIdx.x * blockDim.x + threadIdx.x) >> 5;  // 0..255
    const int lane = threadIdx.x & 31;
    const int d    = w >> 7;
    const int mt   = (w >> 5) & 3;
    const int ht   = w & 31;
    const int lo   = lane & 15;
    const int hi8  = (lane >> 4) ? 8 : 0;    // D-frag M skew
    const int hi16 = (lane >> 4) ? 16 : 0;   // B-frag K skew
    const int arow = mt * 16 + lo;

    v8f acc[4];
#pragma unroll
    for (int g = 0; g < 4; ++g)
#pragma unroll
        for (int r = 0; r < 8; ++r) acc[g][r] = 0.0f;

    // ---- input contribution: X @ Wih^T
    const __bf16* wih_d = wih + (size_t)d * NG * Kx;
    for (int k = 0; k < Kx; k += 32) {
        v16bf a = X_IS_F32 ? a_frag_f32 (Xf + (size_t)arow * ldx, k + hi16)
                           : a_frag_bf16(Xb + (size_t)arow * ldx, k + hi16);
#pragma unroll
        for (int g = 0; g < 4; ++g) {
            const int n = g * H_ + ht * 16 + lo;
            v16bf b = b_frag_bf16(wih_d + (size_t)n * Kx, k + hi16);
            acc[g] = wmma_bf16(a, b, acc[g]);
        }
    }

    // ---- recurrent contribution: Hprev @ Whh^T
    const __bf16* hp    = hprev + (size_t)d * hprev_dir_off;
    const __bf16* whh_d = whh   + (size_t)d * NG * H_;
    for (int k = 0; k < H_; k += 32) {
        v16bf a = a_frag_bf16(hp + (size_t)arow * hprev_ld, k + hi16);
#pragma unroll
        for (int g = 0; g < 4; ++g) {
            const int n = g * H_ + ht * 16 + lo;
            v16bf b = b_frag_bf16(whh_d + (size_t)n * H_, k + hi16);
            acc[g] = wmma_bf16(a, b, acc[g]);
        }
    }

    // ---- fused LSTM cell (gate order i, f, g, o)
    const int hidx = ht * 16 + lo;
    const float bi = bias[d * NG + 0 * H_ + hidx];
    const float bf = bias[d * NG + 1 * H_ + hidx];
    const float bg = bias[d * NG + 2 * H_ + hidx];
    const float bo = bias[d * NG + 3 * H_ + hidx];
    float*  cd = cstate + (size_t)d * B_ * H_;
    __bf16* hn = hnew   + (size_t)d * hnew_dir_off;
#pragma unroll
    for (int r = 0; r < 8; ++r) {
        const int m  = mt * 16 + r + hi8;
        const float ig = sigmoidf_(acc[0][r] + bi);
        const float fg = sigmoidf_(acc[1][r] + bf);
        const float gg = tanhf    (acc[2][r] + bg);
        const float og = sigmoidf_(acc[3][r] + bo);
        const float c  = cd[(size_t)m * H_ + hidx];
        const float c2 = fg * c + ig * gg;
        const float h2 = og * tanhf(c2);
        cd[(size_t)m * H_ + hidx]        = c2;
        hn[(size_t)m * hnew_ld + hidx]   = (__bf16)h2;
    }
}

// ---------------------------------------------------------------------------
// Fused heads: mu = z@Wmu^T+bmu ; lv = z@Wvar^T+bvar ; out = mu+eps*exp(.5*lv)
// 1024 waves; each wave computes the mu- and lv-tile for one (mt, nt),
// sharing the z A-fragment between the two WMMA chains.
// ---------------------------------------------------------------------------
__global__ __launch_bounds__(256)
void head_kernel(const __bf16* __restrict__ z,     // [B][ZD] bf16
                 const float* __restrict__ Wmu,  const float* __restrict__ bmu,
                 const float* __restrict__ Wvar, const float* __restrict__ bvar,
                 const float* __restrict__ eps,  float* __restrict__ out)
{
    const int w    = (blockIdx.x * blockDim.x + threadIdx.x) >> 5;  // 0..1023
    const int lane = threadIdx.x & 31;
    const int mt   = w >> 8;     // 0..3
    const int nt   = w & 255;    // 0..255
    const int lo   = lane & 15;
    const int hi8  = (lane >> 4) ? 8 : 0;
    const int hi16 = (lane >> 4) ? 16 : 0;
    const int arow = mt * 16 + lo;
    const int n    = nt * 16 + lo;

    v8f amu, alv;
#pragma unroll
    for (int r = 0; r < 8; ++r) { amu[r] = 0.0f; alv[r] = 0.0f; }

    for (int k = 0; k < ZD; k += 32) {
        v16bf a  = a_frag_bf16(z + (size_t)arow * ZD, k + hi16);
        v16bf b0 = b_frag_f32(Wmu  + (size_t)n * ZD, k + hi16);
        v16bf b1 = b_frag_f32(Wvar + (size_t)n * ZD, k + hi16);
        amu = wmma_bf16(a, b0, amu);
        alv = wmma_bf16(a, b1, alv);
    }
    const float bm = bmu[n], bv = bvar[n];
#pragma unroll
    for (int r = 0; r < 8; ++r) {
        const int m  = mt * 16 + r + hi8;
        const float mu = amu[r] + bm;
        const float lv = alv[r] + bv;
        out[(size_t)m * ZD + n] = mu + eps[(size_t)m * ZD + n] * __expf(0.5f * lv);
    }
}

// ---------------------------------------------------------------------------
// Small utility kernels
// ---------------------------------------------------------------------------
__global__ void cvt_bf16_kernel(const float* __restrict__ src, __bf16* __restrict__ dst, int n) {
    int i = blockIdx.x * blockDim.x + threadIdx.x;
    if (i < n) dst[i] = (__bf16)src[i];
}

__global__ void bias_add_kernel(const float* __restrict__ a, const float* __restrict__ b,
                                float* __restrict__ dst, int n) {
    int i = blockIdx.x * blockDim.x + threadIdx.x;
    if (i < n) dst[i] = a[i] + b[i];
}

__global__ void zero_kernel(unsigned int* __restrict__ p, int nwords) {
    int i = blockIdx.x * blockDim.x + threadIdx.x;
    if (i < nwords) p[i] = 0u;
}

// z[b, d*1024 + j]: j<512 -> h_d[b][j], else c_d[b][j-512]; d in [l0f,l0r,l1f,l1r]
__global__ void gather_z_kernel(const __bf16* __restrict__ X1f,  // [B][1024] = [h0f|h0r]
                                const __bf16* __restrict__ H1f,  // [2][B][H]
                                const float*  __restrict__ c0,   // [2][B][H]
                                const float*  __restrict__ c1,   // [2][B][H]
                                __bf16* __restrict__ z) {
    int idx = blockIdx.x * blockDim.x + threadIdx.x;
    if (idx >= B_ * ZD) return;
    int b = idx >> 12, col = idx & 4095;
    int d = col >> 10, j = col & 1023;
    __bf16 v;
    if (j < H_) {
        if (d == 0)      v = X1f[b * 1024 + j];
        else if (d == 1) v = X1f[b * 1024 + H_ + j];
        else             v = H1f[(size_t)(d - 2) * B_ * H_ + b * H_ + j];
    } else {
        int jj = j - H_;
        float c = (d < 2) ? c0[(size_t)d * B_ * H_ + b * H_ + jj]
                          : c1[(size_t)(d - 2) * B_ * H_ + b * H_ + jj];
        v = (__bf16)c;
    }
    z[idx] = v;
}

// ---------------------------------------------------------------------------
extern "C" void kernel_launch(void* const* d_in, const int* in_sizes, int n_in,
                              void* d_out, int out_size, void* d_ws, size_t ws_size,
                              hipStream_t stream) {
    const float* input = (const float*)d_in[0];
    const float* eps   = (const float*)d_in[1];
    const float* w_ih0 = (const float*)d_in[2];
    const float* w_hh0 = (const float*)d_in[3];
    const float* b_ih0 = (const float*)d_in[4];
    const float* b_hh0 = (const float*)d_in[5];
    const float* w_ih1 = (const float*)d_in[6];
    const float* w_hh1 = (const float*)d_in[7];
    const float* b_ih1 = (const float*)d_in[8];
    const float* b_hh1 = (const float*)d_in[9];
    const float* W_mu  = (const float*)d_in[10];
    const float* b_mu  = (const float*)d_in[11];
    const float* W_var = (const float*)d_in[12];
    const float* b_var = (const float*)d_in[13];
    float* out = (float*)d_out;
    (void)in_sizes; (void)n_in; (void)out_size; (void)ws_size;

    char* ws = (char*)d_ws;
    size_t off = 0;
    auto wsalloc = [&](size_t bytes) {
        void* p = ws + off;
        off = (off + bytes + 255) & ~(size_t)255;
        return p;
    };
    __bf16* wih0  = (__bf16*)wsalloc((size_t)2 * NG * E_      * 2);
    __bf16* whh0  = (__bf16*)wsalloc((size_t)2 * NG * H_      * 2);
    __bf16* wih1  = (__bf16*)wsalloc((size_t)2 * NG * (2*H_)  * 2);
    __bf16* whh1  = (__bf16*)wsalloc((size_t)2 * NG * H_      * 2);
    float*  bias0 = (float*) wsalloc((size_t)2 * NG * 4);
    float*  bias1 = (float*) wsalloc((size_t)2 * NG * 4);
    char*   state_base = ws + off;
    __bf16* X1    = (__bf16*)wsalloc((size_t)2 * B_ * 1024 * 2);      // ping-pong [h0f|h0r]
    __bf16* H1    = (__bf16*)wsalloc((size_t)2 * 2 * B_ * H_ * 2);    // ping-pong [h1f][h1r]
    float*  c0    = (float*) wsalloc((size_t)2 * B_ * H_ * 4);        // [c0f][c0r]
    float*  c1    = (float*) wsalloc((size_t)2 * B_ * H_ * 4);        // [c1f][c1r]
    size_t  state_bytes = (size_t)((ws + off) - state_base);
    __bf16* zbuf  = (__bf16*)wsalloc((size_t)B_ * ZD * 2);

    // -- one-time prep (re-run every launch for determinism) --
    auto cvt = [&](const float* s, __bf16* dst, int n) {
        cvt_bf16_kernel<<<(n + 255) / 256, 256, 0, stream>>>(s, dst, n);
    };
    cvt(w_ih0, wih0, 2 * NG * E_);
    cvt(w_hh0, whh0, 2 * NG * H_);
    cvt(w_ih1, wih1, 2 * NG * 2 * H_);
    cvt(w_hh1, whh1, 2 * NG * H_);
    bias_add_kernel<<<(2 * NG + 255) / 256, 256, 0, stream>>>(b_ih0, b_hh0, bias0, 2 * NG);
    bias_add_kernel<<<(2 * NG + 255) / 256, 256, 0, stream>>>(b_ih1, b_hh1, bias1, 2 * NG);
    int zwords = (int)(state_bytes / 4);
    zero_kernel<<<(zwords + 255) / 256, 256, 0, stream>>>((unsigned int*)state_base, zwords);

    // -- sequential recurrence: 255 steps, 2 layer-kernels each --
    for (int t = 0; t < S_ - 1; ++t) {
        const int rp = t & 1, wp = rp ^ 1;
        const float* xt = input + (size_t)t * B_ * E_;
        __bf16* X1r = X1 + (size_t)rp * B_ * 1024;
        __bf16* X1w = X1 + (size_t)wp * B_ * 1024;
        __bf16* H1r = H1 + (size_t)rp * 2 * B_ * H_;
        __bf16* H1w = H1 + (size_t)wp * 2 * B_ * H_;

        // layer 0: X = raw input (f32), h-state lives in the X1 concat buffer
        lstm_step_kernel<true><<<32, 256, 0, stream>>>(
            xt, nullptr, E_, E_,
            X1r, /*hprev_dir_off=*/H_, /*hprev_ld=*/1024,
            wih0, whh0, bias0, c0,
            X1w, /*hnew_dir_off=*/H_, /*hnew_ld=*/1024);

        // layer 1: X = layer-0 output of this step (bf16, already concatenated)
        lstm_step_kernel<false><<<32, 256, 0, stream>>>(
            nullptr, X1w, 1024, 1024,
            H1r, /*hprev_dir_off=*/B_ * H_, /*hprev_ld=*/H_,
            wih1, whh1, bias1, c1,
            H1w, /*hnew_dir_off=*/B_ * H_, /*hnew_ld=*/H_);
    }

    // -- final states at parity (S_-1)&1 --
    const int fp = (S_ - 1) & 1;
    gather_z_kernel<<<(B_ * ZD + 255) / 256, 256, 0, stream>>>(
        X1 + (size_t)fp * B_ * 1024, H1 + (size_t)fp * 2 * B_ * H_, c0, c1, zbuf);

    head_kernel<<<128, 256, 0, stream>>>(zbuf, W_mu, b_mu, W_var, b_var, eps, out);
}
